// TopK_PointNetPP_54700703482022
// MI455X (gfx1250) — matrix operand order repaired and checked by reading, not compiled, not run on hardware
//
#include <hip/hip_runtime.h>
#include <hip/hip_bf16.h>

typedef __attribute__((ext_vector_type(16))) __bf16 v16bf;
typedef __attribute__((ext_vector_type(8)))  float  v8f;

#define NPTS    32768
#define BGRAPH  8
#define NPG_    4096
#define KPG     2048
#define MQ      16384           // BGRAPH*KPG
#define OUT1    2097152         // MQ*128
#define OUT2    2146304         // OUT1 + MQ*3

__device__ __forceinline__ unsigned short f2bf(float f) {
    unsigned u = __float_as_uint(f);
    u += 0x7fffu + ((u >> 16) & 1u);
    return (unsigned short)(u >> 16);
}

__device__ __forceinline__ v16bf load_frag2(const unsigned short* p0, const unsigned short* p1) {
    union { uint4 u[2]; v16bf v; } t;
    t.u[0] = *(const uint4*)p0;
    t.u[1] = *(const uint4*)p1;
    return t.v;
}

// ---------------------------------------------------------------------------
// Kernel 1: per-graph bitonic top-K by score (descending, ties -> lower index)
// ---------------------------------------------------------------------------
__global__ __launch_bounds__(1024) void topk_kernel(const float* __restrict__ score,
                                                    const float* __restrict__ pos,
                                                    float* __restrict__ outp,
                                                    int* __restrict__ idxbuf) {
    __shared__ unsigned long long keys[NPG_];
    const int tid = threadIdx.x;
    const int b = blockIdx.x;
    for (int i = tid; i < NPG_; i += 1024) {
        unsigned sb = __float_as_uint(score[b * NPG_ + i]);
        keys[i] = ((unsigned long long)sb << 32) | (unsigned)(~i);
    }
    __syncthreads();
    for (int k = 2; k <= NPG_; k <<= 1) {
        for (int j = k >> 1; j > 0; j >>= 1) {
            for (int i = tid; i < NPG_; i += 1024) {
                int ixj = i ^ j;
                if (ixj > i) {
                    unsigned long long a = keys[i], c = keys[ixj];
                    bool ascend = ((i & k) == 0);
                    bool sw = ascend ? (a > c) : (a < c);
                    if (sw) { keys[i] = c; keys[ixj] = a; }
                }
            }
            __syncthreads();
        }
    }
    for (int t = tid; t < KPG; t += 1024) {
        unsigned long long key = keys[NPG_ - 1 - t];      // descending
        unsigned lo = (unsigned)key;
        int loc = (int)(~lo);
        int g = b * NPG_ + loc;
        int m = b * KPG + t;
        idxbuf[m] = g;
        outp[OUT1 + m * 3 + 0] = pos[g * 3 + 0];
        outp[OUT1 + m * 3 + 1] = pos[g * 3 + 1];
        outp[OUT1 + m * 3 + 2] = pos[g * 3 + 2];
        outp[OUT2 + m] = (float)b;
    }
}

// ---------------------------------------------------------------------------
// Kernel 2: ball query — first 16 points (by index) within radius, wave32 scan
// ---------------------------------------------------------------------------
__global__ __launch_bounds__(256) void ballquery_kernel(const float* __restrict__ pos,
                                                        const int* __restrict__ idxbuf,
                                                        int* __restrict__ nbr,
                                                        unsigned* __restrict__ vmask) {
    __shared__ float lx[NPG_], ly[NPG_], lz[NPG_];
    const int tid = threadIdx.x;
    const int mbase = blockIdx.x * 8;
    const int b = mbase / KPG;
    const int gbase = b * NPG_;
    for (int i = tid; i < NPG_; i += 256) {
        lx[i] = pos[(gbase + i) * 3 + 0];
        ly[i] = pos[(gbase + i) * 3 + 1];
        lz[i] = pos[(gbase + i) * 3 + 2];
    }
    __syncthreads();
    const int w = tid >> 5, lane = tid & 31;
    const int m = mbase + w;
    const int g = idxbuf[m];
    const int qloc = g - gbase;
    const float qx = lx[qloc], qy = ly[qloc], qz = lz[qloc];
    const float R2 = 0.01f;
    int cnt = 0;
    unsigned vbits = 0;
    const unsigned lt = (lane == 0) ? 0u : (0xffffffffu >> (32 - lane));
    for (int base = 0; base < NPG_ && cnt < 16; base += 32) {
        int p = base + lane;
        float dx = lx[p] - qx, dy = ly[p] - qy, dz = lz[p] - qz;
        bool within = (dx * dx + dy * dy + dz * dz) <= R2;
        unsigned mask = (unsigned)__ballot(within);
        if (mask) {
            int slot = cnt + __popc(mask & lt);
            unsigned mybit = 0;
            if (within && slot < 16) {
                int node = gbase + p;
                nbr[m * 16 + slot] = node;
                if (node != m) mybit = 1u << slot;   // PyG self-loop quirk: node idx vs row idx
            }
            #pragma unroll
            for (int off = 16; off; off >>= 1) mybit |= __shfl_xor(mybit, off, 32);
            vbits |= mybit;
            cnt += __popc(mask);
            if (cnt > 16) cnt = 16;
        }
    }
    if (lane < 16 && lane >= cnt) nbr[m * 16 + lane] = gbase;   // safe dummy (masked)
    if (lane == 0) vmask[m] = vbits;
}

// ---------------------------------------------------------------------------
// Kernel 3: x -> bf16
// ---------------------------------------------------------------------------
__global__ void convx_kernel(const float* __restrict__ x, unsigned short* __restrict__ xbf) {
    int i = blockIdx.x * 256 + threadIdx.x;
    if (i < NPTS * 64) xbf[i] = f2bf(x[i]);
}

// ---------------------------------------------------------------------------
// Kernel 4: pre-swizzle weights into WMMA B-fragment layout (per-lane 16 bf16)
//   frag(c,nt): lane l holds col n = nt*16+(l&15), K = c*32 + ((l>=16)?8:0) + ((j<8)?j:j+8)
// ---------------------------------------------------------------------------
__global__ void wfrag_kernel(const float* __restrict__ w1, const float* __restrict__ w2,
                             const float* __restrict__ w3, unsigned short* __restrict__ wf) {
    int f = blockIdx.x * 256 + threadIdx.x;
    if (f >= 18432) return;
    float v;
    if (f < 6144) {               // w1: [96(pad from 67)][64], 3 chunks x 4 ntiles
        int e = f, c = e / 2048, rem = e % 2048, nt = rem / 512, l = (rem % 512) / 16, j = rem % 16;
        int n = nt * 16 + (l & 15);
        int kk = c * 32 + ((l >= 16) ? 8 : 0) + ((j < 8) ? j : j + 8);
        v = (kk < 67) ? w1[kk * 64 + n] : 0.f;
    } else if (f < 10240) {       // w2: [64][64], 2 chunks x 4 ntiles
        int e = f - 6144, c = e / 2048, rem = e % 2048, nt = rem / 512, l = (rem % 512) / 16, j = rem % 16;
        int n = nt * 16 + (l & 15);
        int kk = c * 32 + ((l >= 16) ? 8 : 0) + ((j < 8) ? j : j + 8);
        v = w2[kk * 64 + n];
    } else {                      // w3: [64][128], 2 chunks x 8 ntiles
        int e = f - 10240, c = e / 4096, rem = e % 4096, nt = rem / 512, l = (rem % 512) / 16, j = rem % 16;
        int n = nt * 16 + (l & 15);
        int kk = c * 32 + ((l >= 16) ? 8 : 0) + ((j < 8) ? j : j + 8);
        v = w3[kk * 128 + n];
    }
    wf[f] = f2bf(v);
}

// ---------------------------------------------------------------------------
// Kernel 5: WMMA MLP (67->64->64->128) + masked max-aggregate
//   block = 16 queries (272 rows = 17 tiles of 16), 4 waves round-robin tiles
// ---------------------------------------------------------------------------
__global__ __launch_bounds__(128) void mlp_kernel(const float* __restrict__ pos,
                                                  const int* __restrict__ idxbuf,
                                                  const int* __restrict__ nbr,
                                                  const unsigned* __restrict__ vmask,
                                                  const unsigned short* __restrict__ xbf,
                                                  const unsigned short* __restrict__ wf,
                                                  const float* __restrict__ b1,
                                                  const float* __restrict__ b2,
                                                  const float* __restrict__ b3,
                                                  float* __restrict__ outp) {
    __shared__ __align__(16) unsigned short P[4][16 * 96];   // A0 tile, reused for H2
    __shared__ __align__(16) unsigned short Q[4][16 * 64];   // H1 tile
    __shared__ float agg[16 * 128];
    __shared__ int rowq[4][16];
    __shared__ int rowok[4][16];

    const int tid = threadIdx.x;
    const int w = tid >> 5;
    const int lane = tid & 31;
    const int qbase = blockIdx.x * 16;

    for (int i = tid; i < 16 * 128; i += 128) agg[i] = -3.4e38f;
    __syncthreads();

    const unsigned short* w1f = wf;
    const unsigned short* w2f = wf + 6144;
    const unsigned short* w3f = wf + 10240;

    const int arow = lane & 15;
    const int khi  = (lane >> 4) << 3;      // A/C: 0 or 8
    const int ccol = lane & 15;
    const int rhi  = (lane >> 4) << 3;

    for (int t = w; t < 17; t += 4) {
        // ---- stage A0: [16 rows x 96 cols] bf16 (64 x-features | 3 rel | zero pad)
        {
            int r = lane >> 1;
            int half = lane & 1;
            int rowlocal = t * 16 + r;
            int qq = rowlocal / 17;
            int s  = rowlocal % 17;
            int m  = qbase + qq;
            int g  = idxbuf[m];
            int node, ok;
            if (s < 16) { node = nbr[m * 16 + s]; ok = (int)((vmask[m] >> s) & 1u); }
            else        { node = m;               ok = 1; }
            const uint4* src = (const uint4*)(xbf + node * 64 + half * 32);
            uint4* dst = (uint4*)(&P[w][r * 96 + half * 32]);
            dst[0] = src[0]; dst[1] = src[1]; dst[2] = src[2]; dst[3] = src[3];
            if (half == 0) {
                float rx = pos[node * 3 + 0] - pos[g * 3 + 0];
                float ry = pos[node * 3 + 1] - pos[g * 3 + 1];
                float rz = pos[node * 3 + 2] - pos[g * 3 + 2];
                unsigned* pu = (unsigned*)(&P[w][r * 96 + 64]);
                pu[0] = (unsigned)f2bf(rx) | ((unsigned)f2bf(ry) << 16);
                pu[1] = (unsigned)f2bf(rz);
                #pragma unroll
                for (int z = 2; z < 16; ++z) pu[z] = 0u;
                rowq[w][r]  = qq;
                rowok[w][r] = ok;
            }
        }
        __builtin_amdgcn_wave_barrier();

        // ---- L1: [16x96]x[96x64] -> H1 (relu, bf16)
        v8f acc[8];
        #pragma unroll
        for (int n = 0; n < 4; ++n)
            #pragma unroll
            for (int v = 0; v < 8; ++v) acc[n][v] = 0.f;
        {
            const unsigned short* Abase = &P[w][arow * 96];
            #pragma unroll
            for (int c = 0; c < 3; ++c) {
                v16bf af = load_frag2(Abase + c * 32 + khi, Abase + c * 32 + khi + 16);
                #pragma unroll
                for (int n = 0; n < 4; ++n) {
                    const unsigned short* fb = w1f + (c * 4 + n) * 512 + lane * 16;
                    v16bf bf = load_frag2(fb, fb + 8);
                    acc[n] = __builtin_amdgcn_wmma_f32_16x16x32_bf16(
                        false, af, false, bf, (short)0, acc[n], false, false);
                }
            }
        }
        #pragma unroll
        for (int n = 0; n < 4; ++n) {
            float bv = b1[n * 16 + ccol];
            #pragma unroll
            for (int v = 0; v < 8; ++v) {
                float h = acc[n][v] + bv;
                h = h > 0.f ? h : 0.f;
                Q[w][(rhi + v) * 64 + n * 16 + ccol] = f2bf(h);
            }
        }
        __builtin_amdgcn_wave_barrier();

        // ---- L2: [16x64]x[64x64] -> H2 (relu, bf16, stored over A0 region)
        #pragma unroll
        for (int n = 0; n < 4; ++n)
            #pragma unroll
            for (int v = 0; v < 8; ++v) acc[n][v] = 0.f;
        {
            const unsigned short* Abase = &Q[w][arow * 64];
            #pragma unroll
            for (int c = 0; c < 2; ++c) {
                v16bf af = load_frag2(Abase + c * 32 + khi, Abase + c * 32 + khi + 16);
                #pragma unroll
                for (int n = 0; n < 4; ++n) {
                    const unsigned short* fb = w2f + (c * 4 + n) * 512 + lane * 16;
                    v16bf bf = load_frag2(fb, fb + 8);
                    acc[n] = __builtin_amdgcn_wmma_f32_16x16x32_bf16(
                        false, af, false, bf, (short)0, acc[n], false, false);
                }
            }
        }
        unsigned short* H2 = &P[w][0];
        #pragma unroll
        for (int n = 0; n < 4; ++n) {
            float bv = b2[n * 16 + ccol];
            #pragma unroll
            for (int v = 0; v < 8; ++v) {
                float h = acc[n][v] + bv;
                h = h > 0.f ? h : 0.f;
                H2[(rhi + v) * 64 + n * 16 + ccol] = f2bf(h);
            }
        }
        __builtin_amdgcn_wave_barrier();

        // ---- L3: [16x64]x[64x128] + bias, masked max into agg
        v8f a3[8];
        #pragma unroll
        for (int n = 0; n < 8; ++n)
            #pragma unroll
            for (int v = 0; v < 8; ++v) a3[n][v] = 0.f;
        {
            const unsigned short* Abase = H2 + arow * 64;
            #pragma unroll
            for (int c = 0; c < 2; ++c) {
                v16bf af = load_frag2(Abase + c * 32 + khi, Abase + c * 32 + khi + 16);
                #pragma unroll
                for (int n = 0; n < 8; ++n) {
                    const unsigned short* fb = w3f + (c * 8 + n) * 512 + lane * 16;
                    v16bf bf = load_frag2(fb, fb + 8);
                    a3[n] = __builtin_amdgcn_wmma_f32_16x16x32_bf16(
                        false, af, false, bf, (short)0, a3[n], false, false);
                }
            }
        }
        #pragma unroll
        for (int n = 0; n < 8; ++n) {
            float bv = b3[n * 16 + ccol];
            #pragma unroll
            for (int v = 0; v < 8; ++v) {
                int r = rhi + v;
                if (rowok[w][r]) {
                    __hip_atomic_fetch_max(&agg[rowq[w][r] * 128 + n * 16 + ccol],
                                           a3[n][v] + bv,
                                           __ATOMIC_RELAXED, __HIP_MEMORY_SCOPE_WORKGROUP);
                }
            }
        }
        __builtin_amdgcn_wave_barrier();
    }

    __syncthreads();
    for (int i = tid; i < 16 * 128; i += 128)
        outp[(size_t)qbase * 128 + i] = agg[i];
}

// ---------------------------------------------------------------------------
extern "C" void kernel_launch(void* const* d_in, const int* in_sizes, int n_in,
                              void* d_out, int out_size, void* d_ws, size_t ws_size,
                              hipStream_t stream) {
    const float* x     = (const float*)d_in[0];
    const float* score = (const float*)d_in[1];
    const float* pos   = (const float*)d_in[2];
    const float* w1    = (const float*)d_in[4];
    const float* b1    = (const float*)d_in[5];
    const float* w2    = (const float*)d_in[6];
    const float* b2    = (const float*)d_in[7];
    const float* w3    = (const float*)d_in[8];
    const float* b3    = (const float*)d_in[9];
    float* outp = (float*)d_out;

    int* idxbuf = (int*)d_ws;                                  // MQ ints
    int* nbr = idxbuf + MQ;                                    // MQ*16 ints
    unsigned* vmask = (unsigned*)(nbr + MQ * 16);              // MQ
    unsigned short* xbf = (unsigned short*)(vmask + MQ);       // NPTS*64 bf16
    unsigned short* wf = xbf + (size_t)NPTS * 64;              // 18432 bf16

    topk_kernel<<<BGRAPH, 1024, 0, stream>>>(score, pos, outp, idxbuf);
    convx_kernel<<<(NPTS * 64) / 256, 256, 0, stream>>>(x, xbf);
    wfrag_kernel<<<(18432 + 255) / 256, 256, 0, stream>>>(w1, w2, w3, wf);
    ballquery_kernel<<<MQ / 8, 256, 0, stream>>>(pos, idxbuf, nbr, vmask);
    mlp_kernel<<<MQ / 16, 128, 0, stream>>>(pos, idxbuf, nbr, vmask, xbf, wf,
                                            b1, b2, b3, outp);
}
